// RegionProposalNetwork_47622597378195
// MI455X (gfx1250) — matrix operand order repaired and checked by reading, not compiled, not run on hardware
//
#include <hip/hip_runtime.h>
#include <math.h>

typedef _Float16 v16h __attribute__((ext_vector_type(16)));
typedef _Float16 v8h  __attribute__((ext_vector_type(8)));
typedef float    v8f  __attribute__((ext_vector_type(8)));

#define NB 4
#define CIN 512
#define HSZ 64
#define WSZ 64
#define HW (HSZ*WSZ)          // 4096
#define MTOT (NB*HW)          // 16384
#define COUT 512
#define KTOT (CIN*9)          // 4608
#define NA 9
#define NANCH (HW*NA)         // 36864
#define NSORT 65536
#define NPRE 6000
#define NPOST 300
#define NMS_T 0.7f
#define MIN_SZ 16.0f

// d_out layout (float elements), in reference return order
#define LOC_OFF  0
#define LOC_N    (NB*NANCH*4)
#define SC_OFF   (LOC_OFF+LOC_N)
#define SC_N     (NB*NANCH*2)
#define ROIS_OFF (SC_OFF+SC_N)
#define ROIS_N   (NB*NPOST*4)
#define RIDX_OFF (ROIS_OFF+ROIS_N)
#define RIDX_N   (NB*NPOST)
#define ANC_OFF  (RIDX_OFF+RIDX_N)

__device__ __forceinline__ void anchor_hw(int a, float* ah, float* aw) {
    const float ratios[3] = {0.5f, 1.0f, 2.0f};
    const float scales[3] = {8.0f, 16.0f, 32.0f};
    float r = ratios[a / 3], s = scales[a % 3];
    *ah = 16.0f * s * sqrtf(r);
    *aw = 16.0f * s * sqrtf(1.0f / r);
}

// ---------------- packing kernels ----------------

__global__ void pack_x(const float* __restrict__ x, _Float16* __restrict__ xh) {
    size_t e = (size_t)blockIdx.x * blockDim.x + threadIdx.x;
    if (e >= (size_t)MTOT * CIN) return;
    size_t p = e % HW;
    size_t tmp = e / HW;
    size_t c = tmp % CIN;
    size_t n = tmp / CIN;
    xh[(n * HW + p) * CIN + c] = (_Float16)x[e];   // NCHW f32 -> NHWC f16
}

__global__ void pack_w1(const float* __restrict__ w, _Float16* __restrict__ w1h) {
    size_t e = (size_t)blockIdx.x * blockDim.x + threadIdx.x;
    if (e >= (size_t)COUT * CIN * 9) return;
    int tap = (int)(e % 9);
    size_t tmp = e / 9;
    int cin = (int)(tmp % CIN);
    int cout = (int)(tmp / CIN);
    w1h[(size_t)cout * KTOT + (size_t)tap * CIN + cin] = (_Float16)w[e];
}

__global__ void pack_w2(const float* __restrict__ Wl, const float* __restrict__ bl,
                        const float* __restrict__ Ws, const float* __restrict__ bs,
                        _Float16* __restrict__ w2h, float* __restrict__ b2) {
    int e = blockIdx.x * blockDim.x + threadIdx.x;
    if (e >= 64 * 512) return;
    int c = e % 512, j = e / 512;
    float v = 0.0f;
    if (j < 36)      v = Wl[j * 512 + c];
    else if (j < 54) v = Ws[(j - 36) * 512 + c];
    w2h[e] = (_Float16)v;
    if (c == 0) b2[j] = (j < 36) ? bl[j] : (j < 54 ? bs[j - 36] : 0.0f);
}

// ---------------- 3x3 conv, implicit GEMM with WMMA ----------------
// M=16384 (n*h*w), N=512 (cout), K=4608 (9 taps * 512 cin)
// Double-buffered pipeline: B tiles via global_load_async_to_lds_b128
// (ASYNCcnt), A tiles via registers + ds_store; one barrier per K-step.

__global__ __launch_bounds__(256) void conv3x3_wmma(
    const _Float16* __restrict__ xh, const _Float16* __restrict__ w1h,
    const float* __restrict__ b1, _Float16* __restrict__ hh) {
    __shared__ __align__(16) _Float16 As[2][64][40];
    __shared__ __align__(16) _Float16 Bs[2][64][40];

    const int t = threadIdx.x;
    const int lane = t & 31;
    const int wave = t >> 5;
    const int wm = wave >> 1, wn = wave & 1;
    const int m0 = blockIdx.x * 64;
    const int n0 = blockIdx.y * 64;

    const int lr = t >> 2;           // load row 0..63
    const int lc = (t & 3) * 8;      // load chunk 0,8,16,24 (halfs)

    const int gm = m0 + lr;
    const int nimg = gm >> 12;
    const int pp = gm & 4095;
    const int py = pp >> 6;
    const int px = pp & 63;
    const size_t browBase = (size_t)(n0 + lr) * KTOT;

    const int NIT = 9 * (CIN / 32);  // 144 K-steps of 32

    // async DMA of one 16-byte B chunk into LDS (weights are always in-bounds)
    auto issueB = [&](int it, int buf) {
        const int tap = it >> 4, ck = it & 15;
        unsigned lds = (unsigned)(unsigned long long)&Bs[buf][lr][lc];
        unsigned long long ga =
            (unsigned long long)(w1h + browBase + (size_t)tap * CIN + ck * 32 + lc);
        asm volatile("global_load_async_to_lds_b128 %0, %1, off"
                     :: "v"(lds), "v"(ga) : "memory");
    };
    // zero-padded im2col gather of one 16-byte A chunk into registers
    auto loadA = [&](int it) -> v8h {
        const int tap = it >> 4, ck = it & 15;
        const int ky = tap / 3 - 1, kx = tap % 3 - 1;
        const int sy = py + ky, sx = px + kx;
        v8h av = {};
        if (sy >= 0 && sy < HSZ && sx >= 0 && sx < WSZ)
            av = *(const v8h*)(xh + (size_t)(nimg * HW + sy * WSZ + sx) * CIN
                               + ck * 32 + lc);
        return av;
    };

    v8f c0 = {}; v8f c1 = {};

    // prologue: stage tile 0
    issueB(0, 0);
    { v8h a0 = loadA(0); *(v8h*)(&As[0][lr][lc]) = a0; }
    asm volatile("s_wait_asynccnt 0x0" ::: "memory");
    __syncthreads();

    for (int it = 0; it < NIT; ++it) {
        const int buf = it & 1;
        const bool more = (it + 1 < NIT);
        v8h an = {};
        if (more) { issueB(it + 1, buf ^ 1); an = loadA(it + 1); }

        // A fragment: row M = wm*16 + lane%16; halves 0-7 at K=ko..ko+7, 8-15 at K=ko+16..
        const int arow = wm * 16 + (lane & 15);
        const int ko = (lane & 16) ? 8 : 0;
        v16h a;
        *(v8h*)&a       = *(const v8h*)(&As[buf][arow][ko]);
        *((v8h*)&a + 1) = *(const v8h*)(&As[buf][arow][ko + 16]);

        // B fragments: col N = wn*32 + lane%16 (+16); halves 0-7 at K=kb..kb+7, 8-15 at K=kb+16..
        const int kb = (lane & 16) ? 8 : 0;
        const int bcol = wn * 32 + (lane & 15);
        v16h bf0, bf1;
        *(v8h*)&bf0       = *(const v8h*)(&Bs[buf][bcol][kb]);
        *((v8h*)&bf0 + 1) = *(const v8h*)(&Bs[buf][bcol][kb + 16]);
        *(v8h*)&bf1       = *(const v8h*)(&Bs[buf][bcol + 16][kb]);
        *((v8h*)&bf1 + 1) = *(const v8h*)(&Bs[buf][bcol + 16][kb + 16]);

        c0 = __builtin_amdgcn_wmma_f32_16x16x32_f16(false, a, false, bf0, (short)0, c0, false, false);
        c1 = __builtin_amdgcn_wmma_f32_16x16x32_f16(false, a, false, bf1, (short)0, c1, false, false);

        if (more) *(v8h*)(&As[buf ^ 1][lr][lc]) = an;
        asm volatile("s_wait_asynccnt 0x0" ::: "memory");
        __syncthreads();
    }

    // epilogue: bias + ReLU, store f16 NHWC
    const int hi = (lane & 16) ? 8 : 0;
    const int col0 = n0 + wn * 32 + (lane & 15);
    const float bb0 = b1[col0], bb1 = b1[col0 + 16];
    for (int r = 0; r < 8; ++r) {
        const int m = m0 + wm * 16 + r + hi;
        hh[(size_t)m * COUT + col0]      = (_Float16)fmaxf(c0[r] + bb0, 0.0f);
        hh[(size_t)m * COUT + col0 + 16] = (_Float16)fmaxf(c1[r] + bb1, 0.0f);
    }
}

// ---------------- head GEMM (1x1 convs) + fused proposal epilogue ----------------
// M=16384, N=64 (36 locs + 18 scores + pad), K=512

__global__ __launch_bounds__(256) void head_wmma(
    const _Float16* __restrict__ hh, const _Float16* __restrict__ w2h,
    const float* __restrict__ b2, float* __restrict__ out,
    float* __restrict__ boxes, float* __restrict__ scbuf,
    const int* __restrict__ pimh, const int* __restrict__ pimw) {
    __shared__ __align__(16) _Float16 As[64][40];
    __shared__ __align__(16) _Float16 Bs[64][40];
    __shared__ float Cs[64][68];

    const int t = threadIdx.x;
    const int lane = t & 31;
    const int wave = t >> 5;
    const int wm = wave >> 1, wn = wave & 1;
    const int m0 = blockIdx.x * 64;

    const int lr = t >> 2;
    const int lc = (t & 3) * 8;
    const size_t arowg = (size_t)(m0 + lr) * COUT;
    const size_t browg = (size_t)lr * CIN;

    v8f c0 = {}; v8f c1 = {};

    for (int ck = 0; ck < CIN / 32; ++ck) {
        *(v8h*)(&As[lr][lc]) = *(const v8h*)(hh + arowg + (size_t)ck * 32 + lc);
        *(v8h*)(&Bs[lr][lc]) = *(const v8h*)(w2h + browg + (size_t)ck * 32 + lc);
        __syncthreads();

        const int arow = wm * 16 + (lane & 15);
        const int ko = (lane & 16) ? 8 : 0;
        v16h a;
        *(v8h*)&a       = *(const v8h*)(&As[arow][ko]);
        *((v8h*)&a + 1) = *(const v8h*)(&As[arow][ko + 16]);

        const int kb = (lane & 16) ? 8 : 0;
        const int bcol = wn * 32 + (lane & 15);
        v16h bf0, bf1;
        *(v8h*)&bf0       = *(const v8h*)(&Bs[bcol][kb]);
        *((v8h*)&bf0 + 1) = *(const v8h*)(&Bs[bcol][kb + 16]);
        *(v8h*)&bf1       = *(const v8h*)(&Bs[bcol + 16][kb]);
        *((v8h*)&bf1 + 1) = *(const v8h*)(&Bs[bcol + 16][kb + 16]);

        c0 = __builtin_amdgcn_wmma_f32_16x16x32_f16(false, a, false, bf0, (short)0, c0, false, false);
        c1 = __builtin_amdgcn_wmma_f32_16x16x32_f16(false, a, false, bf1, (short)0, c1, false, false);
        __syncthreads();
    }

    // scatter accumulators + bias into LDS for per-pixel epilogue
    const int hi = (lane & 16) ? 8 : 0;
    const int cc = wn * 32 + (lane & 15);
    for (int r = 0; r < 8; ++r) {
        Cs[wm * 16 + r + hi][cc]      = c0[r] + b2[cc];
        Cs[wm * 16 + r + hi][cc + 16] = c1[r] + b2[cc + 16];
    }
    __syncthreads();

    if (t < 64) {
        const int gm = m0 + t;
        const int nimg = gm >> 12;
        const int pp = gm & 4095;
        const int py = pp >> 6, px = pp & 63;
        const float imh = (float)pimh[0], imw = (float)pimw[0];
        const float cty = 16.0f * py + 8.0f, ctx = 16.0f * px + 8.0f;

        for (int a = 0; a < NA; ++a) {
            float dy = Cs[t][a * 4 + 0], dx = Cs[t][a * 4 + 1];
            float dh = Cs[t][a * 4 + 2], dw = Cs[t][a * 4 + 3];
            float s0 = Cs[t][36 + a * 2], s1 = Cs[t][36 + a * 2 + 1];

            const size_t ai = (size_t)pp * NA + a;
            const size_t li = (size_t)nimg * NANCH + ai;
            out[LOC_OFF + li * 4 + 0] = dy;
            out[LOC_OFF + li * 4 + 1] = dx;
            out[LOC_OFF + li * 4 + 2] = dh;
            out[LOC_OFF + li * 4 + 3] = dw;
            out[SC_OFF + li * 2 + 0] = s0;
            out[SC_OFF + li * 2 + 1] = s1;

            float ah, aw;
            anchor_hw(a, &ah, &aw);
            float cy = dy * ah + cty, cx = dx * aw + ctx;
            float hb = expf(dh) * ah, wb = expf(dw) * aw;
            float y1 = fminf(fmaxf(cy - 0.5f * hb, 0.0f), imh);
            float y2 = fminf(fmaxf(cy + 0.5f * hb, 0.0f), imh);
            float x1 = fminf(fmaxf(cx - 0.5f * wb, 0.0f), imw);
            float x2 = fminf(fmaxf(cx + 0.5f * wb, 0.0f), imw);
            float* bp = boxes + li * 4;
            bp[0] = y1; bp[1] = x1; bp[2] = y2; bp[3] = x2;

            bool valid = ((y2 - y1) >= MIN_SZ) && ((x2 - x1) >= MIN_SZ);
            float fg = 1.0f / (1.0f + expf(s0 - s1));   // softmax class-1 prob
            scbuf[(size_t)nimg * NSORT + ai] = valid ? fg : -INFINITY;
        }
    }
}

// ---------------- sort + NMS ----------------

__global__ void init_sort(float* __restrict__ sc, int* __restrict__ si) {
    int t = blockIdx.x * blockDim.x + threadIdx.x;
    if (t < NB * NSORT) { sc[t] = -INFINITY; si[t] = t & (NSORT - 1); }
}

__global__ void bitonic_pass(float* __restrict__ sc, int* __restrict__ si, int j, int k) {
    int t = blockIdx.x * blockDim.x + threadIdx.x;
    if (t >= NB * (NSORT / 2)) return;
    int img = t >> 15;               // 32768 pairs per image
    int tt = t & 32767;
    int i = 2 * (tt & ~(j - 1)) + (tt & (j - 1));
    int l = i + j;
    float* s = sc + (size_t)img * NSORT;
    int* ix = si + (size_t)img * NSORT;
    bool dirDesc = ((i & k) == 0);
    float a = s[i], b = s[l];
    bool sw = dirDesc ? (a < b) : (a > b);
    if (sw) {
        s[i] = b; s[l] = a;
        int u = ix[i]; ix[i] = ix[l]; ix[l] = u;
    }
}

__global__ __launch_bounds__(1024) void nms_kernel(
    const float* __restrict__ sc, const int* __restrict__ si,
    const float* __restrict__ boxes, float* __restrict__ out) {
    extern __shared__ unsigned char smem[];
    float4* bb = (float4*)smem;                 // NPRE boxes (96 KB in LDS)
    float* ss = (float*)(bb + NPRE);            // NPRE scores
    int* kp = (int*)(ss + NPRE);                // NPRE keep flags
    const int img = blockIdx.x;
    const int t = threadIdx.x;

    for (int i = t; i < NPRE; i += blockDim.x) {
        float s = sc[(size_t)img * NSORT + i];
        int idx = si[(size_t)img * NSORT + i];
        const float* bp = boxes + ((size_t)img * NANCH + idx) * 4;
        bb[i] = make_float4(bp[0], bp[1], bp[2], bp[3]);
        ss[i] = s;
        kp[i] = (s > -1e30f) ? 1 : 0;
    }
    __syncthreads();

    for (int i = 0; i < NPRE; ++i) {
        if (kp[i]) {
            const float4 bi = bb[i];
            const float areai = (bi.z - bi.x) * (bi.w - bi.y);
            for (int j = i + 1 + t; j < NPRE; j += blockDim.x) {
                if (kp[j]) {
                    float4 bj = bb[j];
                    float ty1 = fmaxf(bi.x, bj.x);
                    float tx1 = fmaxf(bi.y, bj.y);
                    float ty2 = fminf(bi.z, bj.z);
                    float tx2 = fminf(bi.w, bj.w);
                    float inter = fmaxf(ty2 - ty1, 0.0f) * fmaxf(tx2 - tx1, 0.0f);
                    float areaj = (bj.z - bj.x) * (bj.w - bj.y);
                    float iou = inter / (areai + areaj - inter + 1e-10f);
                    if (iou > NMS_T) kp[j] = 0;
                }
            }
        }
        __syncthreads();
    }

    if (t == 0) {
        int cnt = 0;
        for (int i = 0; i < NPRE && cnt < NPOST; ++i)
            if (kp[i]) {
                float* rp = out + ROIS_OFF + (size_t)(img * NPOST + cnt) * 4;
                rp[0] = bb[i].x; rp[1] = bb[i].y; rp[2] = bb[i].z; rp[3] = bb[i].w;
                ++cnt;
            }
        for (int i = 0; i < NPRE && cnt < NPOST; ++i)   // pad like top_k over -inf
            if (!kp[i]) {
                float* rp = out + ROIS_OFF + (size_t)(img * NPOST + cnt) * 4;
                rp[0] = bb[i].x; rp[1] = bb[i].y; rp[2] = bb[i].z; rp[3] = bb[i].w;
                ++cnt;
            }
    }
}

__global__ void write_anchors(float* __restrict__ out) {
    int t = blockIdx.x * blockDim.x + threadIdx.x;
    if (t < NANCH) {
        int p = t / NA, a = t % NA;
        int y = p >> 6, x = p & 63;
        float ah, aw;
        anchor_hw(a, &ah, &aw);
        float* o = out + ANC_OFF + (size_t)t * 4;
        o[0] = 16.0f * y + 8.0f - 0.5f * ah;
        o[1] = 16.0f * x + 8.0f - 0.5f * aw;
        o[2] = 16.0f * y + 8.0f + 0.5f * ah;
        o[3] = 16.0f * x + 8.0f + 0.5f * aw;
    }
    if (t < NB * NPOST) ((int*)out)[RIDX_OFF + t] = t / NPOST;
}

// ---------------- launcher ----------------

extern "C" void kernel_launch(void* const* d_in, const int* in_sizes, int n_in,
                              void* d_out, int out_size, void* d_ws, size_t ws_size,
                              hipStream_t stream) {
    const float* x   = (const float*)d_in[0];
    const float* W1  = (const float*)d_in[1];
    const float* b1  = (const float*)d_in[2];
    const float* Ws_ = (const float*)d_in[3];
    const float* bs_ = (const float*)d_in[4];
    const float* Wl_ = (const float*)d_in[5];
    const float* bl_ = (const float*)d_in[6];
    const int* imh   = (const int*)d_in[7];
    const int* imw   = (const int*)d_in[8];
    float* out = (float*)d_out;

    char* ws = (char*)d_ws;
    size_t off = 0;
    _Float16* xh  = (_Float16*)(ws + off); off += (size_t)MTOT * CIN * 2;
    _Float16* w1h = (_Float16*)(ws + off); off += (size_t)COUT * KTOT * 2;
    _Float16* hh  = (_Float16*)(ws + off); off += (size_t)MTOT * COUT * 2;
    _Float16* w2h = (_Float16*)(ws + off); off += (size_t)64 * 512 * 2;
    float* b2     = (float*)(ws + off);    off += 64 * 4;
    float* boxes  = (float*)(ws + off);    off += (size_t)NB * NANCH * 4 * 4;
    float* scb    = (float*)(ws + off);    off += (size_t)NB * NSORT * 4;
    int* sib      = (int*)(ws + off);      off += (size_t)NB * NSORT * 4;

    pack_x<<<(MTOT * CIN + 255) / 256, 256, 0, stream>>>(x, xh);
    pack_w1<<<(COUT * CIN * 9 + 255) / 256, 256, 0, stream>>>(W1, w1h);
    pack_w2<<<(64 * 512 + 255) / 256, 256, 0, stream>>>(Wl_, bl_, Ws_, bs_, w2h, b2);

    conv3x3_wmma<<<dim3(MTOT / 64, COUT / 64), 256, 0, stream>>>(xh, w1h, b1, hh);

    init_sort<<<(NB * NSORT + 255) / 256, 256, 0, stream>>>(scb, sib);
    head_wmma<<<MTOT / 64, 256, 0, stream>>>(hh, w2h, b2, out, boxes, scb, imh, imw);
    write_anchors<<<(NANCH + 255) / 256, 256, 0, stream>>>(out);

    for (int k = 2; k <= NSORT; k <<= 1)
        for (int j = k >> 1; j >= 1; j >>= 1)
            bitonic_pass<<<(NB * (NSORT / 2) + 255) / 256, 256, 0, stream>>>(scb, sib, j, k);

    size_t nms_smem = (size_t)NPRE * sizeof(float4) + (size_t)NPRE * sizeof(float) + (size_t)NPRE * sizeof(int);
    nms_kernel<<<NB, 1024, nms_smem, stream>>>(scb, sib, boxes, out);
}
